// BackpropNetworkWithSTBP_66529043415100
// MI455X (gfx1250) — matrix operand order repaired and checked by reading, not compiled, not run on hardware
//
#include <hip/hip_runtime.h>
#include <hip/hip_bf16.h>

typedef __attribute__((ext_vector_type(16))) _Float16 v16h;
typedef __attribute__((ext_vector_type(8)))  _Float16 v8h;
typedef __attribute__((ext_vector_type(8)))  float    v8f;

// ---------------------------------------------------------------------------
// Problem constants (from reference): B=128, IN=2048, T=32, H=2048, OUT=512
// All three GEMMs share K = 2048.
// ---------------------------------------------------------------------------
#define BB    128
#define KDIM  2048
#define TT    32
#define HH    2048
#define OO    512
#define GRID  64          // persistent blocks (co-resident; one N-strip each)
#define NSTG  (KDIM / 64) // 32 stages of 64-K, quad-buffered, depth-2 pipeline
#define WP64  72          // LDS pitch (f16) per n-row: 144B -> conflict-free ds

// ---------------------------------------------------------------------------
// Elementwise prep kernels
// ---------------------------------------------------------------------------
__global__ void f32_to_f16_kernel(const float* __restrict__ in,
                                  _Float16* __restrict__ out, int n) {
  int i = blockIdx.x * blockDim.x + threadIdx.x;
  if (i < n) out[i] = (_Float16)in[i];
}

// spike_data [B, K, T] f32 -> X [T, B, K] f16  (coalesced writes)
__global__ void pack_spikes_kernel(const float* __restrict__ sd,
                                   _Float16* __restrict__ X) {
  int i = blockIdx.x * blockDim.x + threadIdx.x;   // over T*B*K
  const int total = TT * BB * KDIM;
  if (i >= total) return;
  int k = i % KDIM;
  int b = (i / KDIM) % BB;
  int t = i / (KDIM * BB);
  X[i] = (_Float16)sd[((size_t)b * KDIM + k) * TT + t];
}

__global__ void init_state_kernel(const float* __restrict__ vin,
                                  const float* __restrict__ sin,
                                  float* __restrict__ v,
                                  _Float16* __restrict__ s, int n) {
  int i = blockIdx.x * blockDim.x + threadIdx.x;
  if (i < n) { v[i] = vin[i]; s[i] = (_Float16)sin[i]; }
}

__global__ void zero_f32_kernel(float* __restrict__ p, int n) {
  int i = blockIdx.x * blockDim.x + threadIdx.x;
  if (i < n) p[i] = 0.0f;
}

__global__ void zero_u32_kernel(unsigned* __restrict__ p, int n) {
  int i = blockIdx.x * blockDim.x + threadIdx.x;
  if (i < n) p[i] = 0u;
}

// ---------------------------------------------------------------------------
// Fence-free workgroup barrier for the GEMM inner loop. No loadcnt/dscnt
// drain: LDS ordering is handled explicitly by s_wait_asynccnt, and ds-read
// completion is implied by the dscnt waits feeding the WMMAs.
// ---------------------------------------------------------------------------
__device__ __forceinline__ void lsync() {
  asm volatile("s_barrier_signal -1\n\ts_barrier_wait -1" ::: "memory");
}

// ---------------------------------------------------------------------------
// Device-wide barrier (generation counter). Release fence drains stores +
// global_wb(DEV); acquire fence emits global_inv(DEV) so the read-only WGP$
// cannot serve stale spike/volt lines in the next phase.
// ---------------------------------------------------------------------------
__device__ __forceinline__ void grid_sync(unsigned* cnt, unsigned* gen) {
  __builtin_amdgcn_fence(__ATOMIC_RELEASE, "agent");
  __syncthreads();
  if (threadIdx.x == 0) {
    unsigned g = __hip_atomic_load(gen, __ATOMIC_RELAXED, __HIP_MEMORY_SCOPE_AGENT);
    unsigned prev = __hip_atomic_fetch_add(cnt, 1u, __ATOMIC_ACQ_REL,
                                           __HIP_MEMORY_SCOPE_AGENT);
    if (prev == GRID - 1) {
      __hip_atomic_store(cnt, 0u, __ATOMIC_RELAXED, __HIP_MEMORY_SCOPE_AGENT);
      __hip_atomic_fetch_add(gen, 1u, __ATOMIC_RELEASE, __HIP_MEMORY_SCOPE_AGENT);
    } else {
      while (__hip_atomic_load(gen, __ATOMIC_RELAXED,
                               __HIP_MEMORY_SCOPE_AGENT) == g)
        __builtin_amdgcn_s_sleep(2);
    }
  }
  __syncthreads();
  __builtin_amdgcn_fence(__ATOMIC_ACQUIRE, "agent");
}

// ---------------------------------------------------------------------------
// One fused GEMM+LIF phase. Block = 8 waves; wave w owns M-tile 16*w (whole
// batch per block -> each W strip read from L2 once). Block owns a 32-column
// N strip. W staged 64-K/4KB at a time (one async b128 per thread) into a
// 4-deep LDS ring, depth-2: at stage s, tile s is consumed, tile s+2 issued.
// One barrier round + one s_wait_asynccnt per stage. A operands are software
// pipelined in registers two stages ahead.
// ---------------------------------------------------------------------------
template <bool OUTL>
__device__ __forceinline__ void lif_gemm_phase(
    const _Float16* __restrict__ A,     // [128, 2048] f16 spikes
    const _Float16* __restrict__ W,     // [N, 2048] f16 weights
    const float*    __restrict__ bias,  // [N]
    float*          __restrict__ volt,  // [128, N]
    _Float16*       __restrict__ spk,   // [128, N] (prev in, new out)
    float*          __restrict__ outAcc,// [128, N] (output layer only)
    int N, int strip, _Float16 (&shW)[4][32 * WP64]) {
  const int tid  = threadIdx.x;
  const int lane = tid & 31;
  const int wave = tid >> 5;
  const int sel  = lane >> 4;
  const int l16  = lane & 15;
  const int m0    = wave * 16;
  const int nBase = strip * 32;

  // A operand (contiguous k order): row m0+l16, chunks at sel*16, sel*16+8
  const _Float16* Arow = A + (size_t)(m0 + l16) * KDIM + sel * 16;

  // Staging geometry: tid -> (n-row, 16B eighth of the 128B k-row)
  const int srow = tid >> 3;   // 0..31
  const int sq   = tid & 7;    // 0..7
  const _Float16* gW = W + (size_t)(nBase + srow) * KDIM + sq * 8;
  unsigned ldsAddr[4];
#pragma unroll
  for (int b = 0; b < 4; ++b)
    ldsAddr[b] = (unsigned)(uintptr_t)(&shW[b][srow * WP64 + sq * 8]);

  v8f acc[2];
#pragma unroll
  for (int j = 0; j < 2; ++j) acc[j] = (v8f){0, 0, 0, 0, 0, 0, 0, 0};

  v8h a0[4], a1[4], a2[4], a3[4];  // A ring (2 substeps x 2 chunks each)

  auto issueA = [&](v8h (&ab)[4], int s) {
    const _Float16* p = Arow + (s << 6);
    ab[0] = *(const v8h*)(p);
    ab[1] = *(const v8h*)(p + 8);
    ab[2] = *(const v8h*)(p + 32);
    ab[3] = *(const v8h*)(p + 40);
  };
  auto stageW = [&](unsigned ldsa, int s) {
    const _Float16* g = gW + (s << 6);
    asm volatile("global_load_async_to_lds_b128 %0, %1, off"
                 :: "v"(ldsa), "v"(g) : "memory");
  };
  // Preload all 8 B-vectors (clause), then 4 back-to-back WMMAs.
  auto consume = [&](v8h (&ab)[4], const _Float16* buf) {
    const _Float16* base = buf + l16 * WP64 + sel * 16;
    v8h b00 = *(const v8h*)(base);
    v8h b01 = *(const v8h*)(base + 8);
    v8h b10 = *(const v8h*)(base + 16 * WP64);
    v8h b11 = *(const v8h*)(base + 16 * WP64 + 8);
    v8h c00 = *(const v8h*)(base + 32);
    v8h c01 = *(const v8h*)(base + 40);
    v8h c10 = *(const v8h*)(base + 16 * WP64 + 32);
    v8h c11 = *(const v8h*)(base + 16 * WP64 + 40);
    v16h A0 = __builtin_shufflevector(ab[0], ab[1], 0, 1, 2, 3, 4, 5, 6, 7,
                                      8, 9, 10, 11, 12, 13, 14, 15);
    v16h A1 = __builtin_shufflevector(ab[2], ab[3], 0, 1, 2, 3, 4, 5, 6, 7,
                                      8, 9, 10, 11, 12, 13, 14, 15);
    v16h B0 = __builtin_shufflevector(b00, b01, 0, 1, 2, 3, 4, 5, 6, 7,
                                      8, 9, 10, 11, 12, 13, 14, 15);
    v16h B1 = __builtin_shufflevector(b10, b11, 0, 1, 2, 3, 4, 5, 6, 7,
                                      8, 9, 10, 11, 12, 13, 14, 15);
    v16h C0 = __builtin_shufflevector(c00, c01, 0, 1, 2, 3, 4, 5, 6, 7,
                                      8, 9, 10, 11, 12, 13, 14, 15);
    v16h C1 = __builtin_shufflevector(c10, c11, 0, 1, 2, 3, 4, 5, 6, 7,
                                      8, 9, 10, 11, 12, 13, 14, 15);
    acc[0] = __builtin_amdgcn_wmma_f32_16x16x32_f16(false, A0, false, B0,
                                                    (short)0, acc[0], false, false);
    acc[1] = __builtin_amdgcn_wmma_f32_16x16x32_f16(false, A0, false, B1,
                                                    (short)0, acc[1], false, false);
    acc[0] = __builtin_amdgcn_wmma_f32_16x16x32_f16(false, A1, false, C0,
                                                    (short)0, acc[0], false, false);
    acc[1] = __builtin_amdgcn_wmma_f32_16x16x32_f16(false, A1, false, C1,
                                                    (short)0, acc[1], false, false);
  };
  // One pipeline step: tile s from bufCur; issue tile s+2 into bufNxt.
  auto step = [&](int s, v8h (&aCur)[4], v8h (&aNxt)[4],
                  const _Float16* bufCur, unsigned ldsNxt) {
    if (s == NSTG - 1) asm volatile("s_wait_asynccnt 0x0" ::: "memory");
    else               asm volatile("s_wait_asynccnt 0x1" ::: "memory");
    lsync();  // all waves: tile s landed, and done reading tile s-2's buffer
    if (s + 2 < NSTG) {
      issueA(aNxt, s + 2);
      stageW(ldsNxt, s + 2);
    }
    consume(aCur, bufCur);
  };

  // Prologue: tiles 0 and 1 in flight.
  issueA(a0, 0); stageW(ldsAddr[0], 0);
  issueA(a1, 1); stageW(ldsAddr[1], 1);

#pragma unroll 1
  for (int s = 0; s < NSTG; s += 4) {
    step(s + 0, a0, a2, &shW[0][0], ldsAddr[2]);
    step(s + 1, a1, a3, &shW[1][0], ldsAddr[3]);
    step(s + 2, a2, a0, &shW[2][0], ldsAddr[0]);
    step(s + 3, a3, a1, &shW[3][0], ldsAddr[1]);
  }

  // Epilogue: fused LIF. C/D layout: lane -> n = l16, VGPR r -> m = r + 8*sel.
#pragma unroll
  for (int j = 0; j < 2; ++j) {
    const int n = nBase + j * 16 + l16;
    const float bn = bias[n];
#pragma unroll
    for (int r = 0; r < 8; ++r) {
      const int m = m0 + r + sel * 8;
      const size_t idx = (size_t)m * N + n;
      float v  = volt[idx];
      float sp = (float)spk[idx];
      v = 0.5f * v * (1.0f - sp) + acc[j][r] + bn;
      const float ns = (v > 0.5f) ? 1.0f : 0.0f;
      volt[idx] = v;
      spk[idx]  = (_Float16)ns;
      if (OUTL) outAcc[idx] += ns;
    }
  }
}

// ---------------------------------------------------------------------------
// Persistent kernel: whole T=32 scan (96 GEMM phases) in one launch, with a
// device-wide barrier between phases instead of 96 kernel dispatch/drains.
// ---------------------------------------------------------------------------
__global__ __launch_bounds__(256) void snn_scan_kernel(
    const _Float16* __restrict__ X,    // [T, 128, 2048]
    const _Float16* __restrict__ W0h, const _Float16* __restrict__ W1h,
    const _Float16* __restrict__ Woh,
    const float* __restrict__ b0, const float* __restrict__ b1,
    const float* __restrict__ bo,
    float* __restrict__ v0, float* __restrict__ v1, float* __restrict__ vo,
    _Float16* __restrict__ s0h, _Float16* __restrict__ s1h,
    _Float16* __restrict__ soh,
    float* __restrict__ out, unsigned* cnt, unsigned* gen) {
  __shared__ __align__(16) _Float16 shW[4][32 * WP64];  // 18 KB quad ring
  const int bid = blockIdx.x;

  for (int t = 0; t < TT; ++t) {
    const _Float16* xt = X + (size_t)t * BB * KDIM;
    lif_gemm_phase<false>(xt, W0h, b0, v0, s0h, nullptr, HH, bid, shW);
    grid_sync(cnt, gen);
    lif_gemm_phase<false>(s0h, W1h, b1, v1, s1h, nullptr, HH, bid, shW);
    grid_sync(cnt, gen);
    if (bid < OO / 32)
      lif_gemm_phase<true>(s1h, Woh, bo, vo, soh, out, OO, bid, shW);
    grid_sync(cnt, gen);
  }
}

// ---------------------------------------------------------------------------
// Host launcher
// ---------------------------------------------------------------------------
extern "C" void kernel_launch(void* const* d_in, const int* in_sizes, int n_in,
                              void* d_out, int out_size, void* d_ws,
                              size_t ws_size, hipStream_t stream) {
  (void)in_sizes; (void)n_in; (void)out_size; (void)ws_size;

  const float* spike_data = (const float*)d_in[0];
  const float* h0v = (const float*)d_in[1];
  const float* h0s = (const float*)d_in[2];
  const float* h1v = (const float*)d_in[3];
  const float* h1s = (const float*)d_in[4];
  const float* ovv = (const float*)d_in[5];
  const float* ovs = (const float*)d_in[6];
  const float* W0  = (const float*)d_in[7];
  const float* b0  = (const float*)d_in[8];
  const float* W1  = (const float*)d_in[9];
  const float* b1  = (const float*)d_in[10];
  const float* Wo  = (const float*)d_in[11];
  const float* bo  = (const float*)d_in[12];
  float* out = (float*)d_out;

  // Workspace carve (256B aligned)
  char* ws = (char*)d_ws;
  size_t cur = 0;
  auto carve = [&](size_t bytes) -> void* {
    void* p = ws + cur;
    cur = (cur + bytes + 255) & ~(size_t)255;
    return p;
  };
  _Float16* X   = (_Float16*)carve((size_t)TT * BB * KDIM * 2); // 16 MB
  _Float16* W0h = (_Float16*)carve((size_t)HH * KDIM * 2);      //  8 MB
  _Float16* W1h = (_Float16*)carve((size_t)HH * KDIM * 2);      //  8 MB
  _Float16* Woh = (_Float16*)carve((size_t)OO * KDIM * 2);      //  2 MB
  float*    v0  = (float*)carve((size_t)BB * HH * 4);
  float*    v1  = (float*)carve((size_t)BB * HH * 4);
  float*    vo  = (float*)carve((size_t)BB * OO * 4);
  _Float16* s0h = (_Float16*)carve((size_t)BB * HH * 2);
  _Float16* s1h = (_Float16*)carve((size_t)BB * HH * 2);
  _Float16* soh = (_Float16*)carve((size_t)BB * OO * 2);
  unsigned* bar = (unsigned*)carve(2 * sizeof(unsigned));  // cnt, gen

  const int TPB = 256;
  auto blocks = [&](int n) { return (n + TPB - 1) / TPB; };

  // Prep: weight conversion, spike transpose/pack, state init, zeroing.
  f32_to_f16_kernel<<<blocks(HH * KDIM), TPB, 0, stream>>>(W0, W0h, HH * KDIM);
  f32_to_f16_kernel<<<blocks(HH * KDIM), TPB, 0, stream>>>(W1, W1h, HH * KDIM);
  f32_to_f16_kernel<<<blocks(OO * KDIM), TPB, 0, stream>>>(Wo, Woh, OO * KDIM);
  pack_spikes_kernel<<<blocks(TT * BB * KDIM), TPB, 0, stream>>>(spike_data, X);
  init_state_kernel<<<blocks(BB * HH), TPB, 0, stream>>>(h0v, h0s, v0, s0h, BB * HH);
  init_state_kernel<<<blocks(BB * HH), TPB, 0, stream>>>(h1v, h1s, v1, s1h, BB * HH);
  init_state_kernel<<<blocks(BB * OO), TPB, 0, stream>>>(ovv, ovs, vo, soh, BB * OO);
  zero_f32_kernel<<<blocks(BB * OO), TPB, 0, stream>>>(out, BB * OO);
  zero_u32_kernel<<<1, 32, 0, stream>>>(bar, 2);

  // Entire sequential LIF scan in one persistent launch.
  snn_scan_kernel<<<GRID, TPB, 0, stream>>>(
      X, W0h, W1h, Woh, b0, b1, bo, v0, v1, vo, s0h, s1h, soh, out,
      bar, bar + 1);
}
